// SRUREDC3D_attention_dsv_2_74088185856158
// MI455X (gfx1250) — matrix operand order repaired and checked by reading, not compiled, not run on hardware
//
#include <hip/hip_runtime.h>
#include <hip/hip_bf16.h>

// CDNA5 (gfx1250) fused Conv3d+BN+activations+bidirectional SRU scan.
// One wave = 16-wide w-strip of one (b,h) row. Conv gates via
// v_wmma_f32_16x16x32_bf16 (A = 16 points x K32 patch, B = folded w*gamma,
// C init = beta). Forward scan result parked in LDS; backward scan fuses the
// final sum and writes the output exactly once (~70MB total traffic ->
// ~3us at 23.3 TB/s; conv FLOPs (~7 GF with recompute) are negligible
// against WMMA throughput, so this is purely bandwidth-bound).

typedef __attribute__((ext_vector_type(16))) __bf16 v16bf;
typedef __attribute__((ext_vector_type(8)))  float  v8f;

#define B_    2
#define T_    31
#define H_    128
#define W_    128
#define HW_   (H_*W_)
#define COUT_ 16
#define NG_   6
#define KV_   27   // 3*3*3 taps, padded to K=32

// gfx1250 has a native V_TANH_F32 transcendental (ISA §8.4); use it when the
// toolchain exposes the builtin, else fall back to v_exp+v_rcp.
__device__ __forceinline__ float fast_tanh(float x) {
#if __has_builtin(__builtin_amdgcn_tanhf)
  return __builtin_amdgcn_tanhf(x);
#else
  return 2.0f * __builtin_amdgcn_rcpf(1.0f + __expf(-2.0f * x)) - 1.0f;
#endif
}
__device__ __forceinline__ float fast_sigmoid(float x) {
#if __has_builtin(__builtin_amdgcn_tanhf)
  return 0.5f + 0.5f * __builtin_amdgcn_tanhf(0.5f * x);   // mul+trans+fma
#else
  return __builtin_amdgcn_rcpf(1.0f + __expf(-x));
#endif
}
__device__ __forceinline__ v8f v_sigmoid(v8f x) {
  v8f r;
#pragma unroll
  for (int i = 0; i < 8; ++i) r[i] = fast_sigmoid(x[i]);
  return r;
}
__device__ __forceinline__ v8f v_tanh(v8f x) {
  v8f r;
#pragma unroll
  for (int i = 0; i < 8; ++i) r[i] = fast_tanh(x[i]);
  return r;
}

// D = A(16x32 bf16 patches) * B(32x16 bf16 folded weights) + beta
__device__ __forceinline__ v8f wmma_gate(v16bf a, v16bf bw, float bias) {
  v8f c;
#pragma unroll
  for (int i = 0; i < 8; ++i) c[i] = bias;
  return __builtin_amdgcn_wmma_f32_16x16x32_bf16(
      /*neg_a=*/false, a, /*neg_b=*/false, bw,
      /*c_mod=*/(short)0, c, /*reuse_a=*/false, /*reuse_b=*/false);
}

__global__ __launch_bounds__(128, 1)
void sru_conv3d_fused_kernel(const float* __restrict__ x,
                             const float* __restrict__ cw,
                             const float* __restrict__ gamma,
                             const float* __restrict__ beta,
                             float* __restrict__ out) {
  // forward-scan h_l stash: [wave][t][lane*8 + v]  (127 KB, < 320 KB WGP LDS)
  __shared__ float hl_s[4][T_][256];

  const int lane = threadIdx.x & 31;
  const int wv   = threadIdx.x >> 5;   // wave in block: 0..3
  const int half = lane >> 4;          // wave half (K-split per ISA layout)
  const int ln   = lane & 15;          // row (A) / col (B,D) index

  const int idx = blockIdx.x;
  const int wt  = idx & 1;             // which 64-wide half of the row
  const int h   = (idx >> 1) & (H_ - 1);
  const int b   = idx >> 8;
  const int w0  = wt * 64 + wv * 16;   // this wave's 16-wide strip

  // ---- B matrices: conv_w * gamma folded, zero-padded K=27..31; C bias=beta
  // B layout (32x16 bf16): lane = N (=oc%16), element i -> K = half*8 + i (i<8), 16 + half*8 + (i-8) (i>=8).
  v16bf Bw[NG_];
  float bv[NG_];
#pragma unroll
  for (int g = 0; g < NG_; ++g) {
    const int oc = g * COUT_ + ln;
    const float gm = gamma[oc];
#pragma unroll
    for (int i = 0; i < 16; ++i) {
      const int k = (i < 8) ? (half * 8 + i) : (half * 8 + i + 8);
      const float w = (k < KV_) ? cw[oc * KV_ + k] * gm : 0.0f;
      Bw[g][i] = (__bf16)w;
    }
    bv[g] = beta[oc];
  }

  // ---- per-lane A-patch offsets (H/W bounds folded into sentinel -1).
  // A layout (16x32 bf16): lane half0 holds K{0..7,16..23}, half1 K{8..15,24..31}.
  int aoff[16];
  int atrel[16];
#pragma unroll
  for (int i = 0; i < 16; ++i) {
    const int k = (i < 8) ? (half * 8 + i) : (half * 8 + i + 8);
    if (k < KV_) {
      const int dt = k / 9, r = k % 9, dh = r / 3, dw = r % 3;
      const int hh = h + dh - 1;
      const int ww = w0 + ln + dw - 1;
      aoff[i]  = (hh >= 0 && hh < H_ && ww >= 0 && ww < W_) ? hh * W_ + ww : -1;
      atrel[i] = dt - 1;
    } else {
      aoff[i]  = -1;
      atrel[i] = 0;
    }
  }
  const float* xb = x + (size_t)b * T_ * HW_;

  auto loadA = [&](int t) -> v16bf {
    v16bf a;
#pragma unroll
    for (int i = 0; i < 16; ++i) {
      float v = 0.0f;
      const int tt = t + atrel[i];
      if (aoff[i] >= 0 && (unsigned)tt < (unsigned)T_)
        v = xb[(size_t)tt * HW_ + aoff[i]];
      a[i] = (__bf16)v;
    }
    return a;
  };

  // ---- forward scan: C_0 = 1-ft_0; C_t = ft*C + (1-ft)*wx; hl = rt*C + (1-rt)*xa
  v8f c = {};
  for (int t = 0; t < T_; ++t) {
    if (t + 2 < T_)  // warm L2/L0 for next slices (global_prefetch_b8)
      __builtin_prefetch((const void*)(xb + (size_t)(t + 2) * HW_ + h * W_ + w0), 0, 0);
    const v16bf a = loadA(t);
    const v8f wx = v_tanh   (wmma_gate(a, Bw[0], bv[0]));
    const v8f ft = v_sigmoid(wmma_gate(a, Bw[1], bv[1]));
    const v8f rt = v_sigmoid(wmma_gate(a, Bw[3], bv[3]));
    const v8f xa = v_tanh   (wmma_gate(a, Bw[5], bv[5]));
    c = (t == 0) ? (1.0f - ft) : (ft * c + (1.0f - ft) * wx);
    const v8f hl = rt * c + (1.0f - rt) * xa;
    *(v8f*)(&hl_s[wv][t][lane * 8]) = hl;   // private per-wave region, no barrier
  }

  // ---- backward scan (recompute gates: conv is compute-cheap, kernel is BW-bound)
  v8f c2 = {};
  for (int t = T_ - 1; t >= 0; --t) {
    const v16bf a = loadA(t);
    const v8f wx  = v_tanh   (wmma_gate(a, Bw[0], bv[0]));
    const v8f ft2 = v_sigmoid(wmma_gate(a, Bw[2], bv[2]));
    const v8f rt2 = v_sigmoid(wmma_gate(a, Bw[4], bv[4]));
    const v8f xa  = v_tanh   (wmma_gate(a, Bw[5], bv[5]));
    c2 = (t == T_ - 1) ? (1.0f - ft2) : (ft2 * c2 + (1.0f - ft2) * wx);
    const v8f hr = rt2 * c2 + (1.0f - rt2) * xa;
    const v8f hl = *(const v8f*)(&hl_s[wv][t][lane * 8]);
    const v8f o  = hl + hr;
    // D layout: VGPR v -> M = v + 8*half -> w = w0 + 8*half + v (contiguous, 32B aligned)
    float* op = out + (((size_t)(b * COUT_ + ln) * T_ + t) * HW_) + h * W_ + w0 + half * 8;
    *(v8f*)op = o;
  }
}

extern "C" void kernel_launch(void* const* d_in, const int* in_sizes, int n_in,
                              void* d_out, int out_size, void* d_ws, size_t ws_size,
                              hipStream_t stream) {
  (void)in_sizes; (void)n_in; (void)d_ws; (void)ws_size; (void)out_size;
  const float* x     = (const float*)d_in[0];
  const float* cw    = (const float*)d_in[1];
  const float* gamma = (const float*)d_in[2];
  const float* beta  = (const float*)d_in[3];
  float* out = (float*)d_out;

  dim3 grid(B_ * H_ * 2);   // 512 blocks: (b, h, 64-wide w tile)
  dim3 block(128);          // 4 waves of 32; all lanes active (EXEC all-1s for WMMA)
  hipLaunchKernelGGL(sru_conv3d_fused_kernel, grid, block, 0, stream,
                     x, cw, gamma, beta, out);
}
